// GNN_40724879901230
// MI455X (gfx1250) — compile-verified, matrix operand order checked by reading, not run on hardware
//
#include <hip/hip_runtime.h>
#include <math.h>

typedef __attribute__((ext_vector_type(16))) _Float16 v16h;
typedef __attribute__((ext_vector_type(8)))  float    v8f;

#define D_HID     64
#define TSTEPS    2
#define NEG_SLOPE 0.2f

// ---------------------------------------------------------------- utilities
__device__ __forceinline__ void atomicMaxF(float* addr, float val) {
  // sign-aware bit trick; correct for mixed signs with -inf init
  if (val >= 0.0f) atomicMax((int*)addr, __float_as_int(val));
  else             atomicMin((unsigned int*)addr, __float_as_uint(val));
}

__global__ void k_fill(float* __restrict__ p, float v, long n) {
  long i = (long)blockIdx.x * blockDim.x + threadIdx.x;
  if (i < n) p[i] = v;
}

// ------------------------------------------------------------- WMMA GEMM
// C[nRows x OC] = act( A[nRows x 64] @ B + bias ), fp16 inputs / f32 accum.
// bMode 0: B element (k,n) at B[k*OC + n]   (row-major K x OC)
// bMode 1: B element (k,n) at B[n*64 + k]   (i.e. W^T, W stored [OC x 64])
// B is staged once per block into LDS (f16, column-major sB[n*64+k]) so the
// per-wave fragment reads are contiguous 16B ds loads; one 16x16 tile/wave,
// K=64 as two v_wmma_f32_16x16x32_f16.
__global__ void k_gemm_wmma(const float* __restrict__ A, const float* __restrict__ B,
                            const float* __restrict__ bias, float* __restrict__ C,
                            int nRows, int OC, int bMode, int act) {
  __shared__ __align__(16) _Float16 sB[192 * 64];   // max OC=192 -> 24 KB
  const int tid = threadIdx.x;

  // cooperative stage + convert of B into LDS (column-major)
  for (int idx = tid; idx < OC * D_HID; idx += blockDim.x) {
    const int n = idx >> 6, k = idx & 63;
    const float v = bMode ? B[idx] : B[(size_t)k * OC + n];
    sB[idx] = (_Float16)v;
  }
  __syncthreads();

  const int lane     = tid & 31;
  const int colTiles = OC >> 4;
  const int tile     = blockIdx.x * (blockDim.x >> 5) + (tid >> 5);
  const int totTiles = (nRows >> 4) * colTiles;
  if (tile >= totTiles) return;            // whole-wave uniform exit (after barrier)
  const int rowTile = tile / colTiles;
  const int colTile = tile - rowTile * colTiles;
  const int row0 = rowTile << 4;
  const int col0 = colTile << 4;

  const int mn   = lane & 15;              // A row / B col within tile
  const int koff = (lane >> 4) << 3;       // 0 (lanes 0-15) or 8 (lanes 16-31)
  const _Float16* __restrict__ bcol = sB + ((col0 + mn) << 6);

  v8f acc = {};
#pragma unroll
  for (int kk = 0; kk < D_HID; kk += 32) {
    v16h af, bf;
    // A fragment: row (row0+mn); K = kk + {koff..koff+7, 16+koff..16+koff+7}
    const float* ar = A + (size_t)(row0 + mn) * D_HID + kk;
#pragma unroll
    for (int i = 0; i < 8; ++i) af[i]     = (_Float16)ar[koff + i];
#pragma unroll
    for (int i = 0; i < 8; ++i) af[8 + i] = (_Float16)ar[16 + koff + i];
    // B fragment from LDS: contiguous 8 halves twice (16B-aligned)
    const _Float16* bk = bcol + kk;
#pragma unroll
    for (int i = 0; i < 8; ++i) bf[i]     = bk[koff + i];
#pragma unroll
    for (int i = 0; i < 8; ++i) bf[8 + i] = bk[16 + koff + i];
    acc = __builtin_amdgcn_wmma_f32_16x16x32_f16(false, af, false, bf,
                                                 (short)0, acc, false, false);
  }

  // C/D layout: VGPR j -> M = j (lanes 0-15) or 8+j (lanes 16-31); N = lane&15
  const int   mBase = (lane < 16) ? 0 : 8;
  const float bv    = bias ? bias[col0 + mn] : 0.0f;
#pragma unroll
  for (int j = 0; j < 8; ++j) {
    float v = acc[j] + bv;
    if (act == 1) v = fmaxf(v, 0.0f);
    C[(size_t)(row0 + mBase + j) * OC + col0 + mn] = v;
  }
}

// -------------------------------------------------- attention logit GEMVs
// one wave per node: a_src[n] = h[n,:]·att_src ; a_dst[n] = h[n,:]·att_dst
__global__ void k_attn_dot(const float* __restrict__ h,
                           const float* __restrict__ att_s, const float* __restrict__ att_d,
                           float* __restrict__ a_s, float* __restrict__ a_d, int n) {
  int wave = (int)((blockIdx.x * (long)blockDim.x + threadIdx.x) >> 5);
  int lane = threadIdx.x & 31;
  if (wave >= n) return;
  const float* row = h + (size_t)wave * D_HID;
  float ss = 0.f, sd = 0.f;
#pragma unroll
  for (int d = 0; d < D_HID; d += 32) {
    float v = row[d + lane];
    ss += v * att_s[d + lane];
    sd += v * att_d[d + lane];
  }
#pragma unroll
  for (int o = 16; o > 0; o >>= 1) { ss += __shfl_down(ss, o); sd += __shfl_down(sd, o); }
  if (lane == 0) { a_s[wave] = ss; a_d[wave] = sd; }
}

// ------------------------------------------------------------ edge kernels
__global__ void k_edge_logit(const int* __restrict__ src, const int* __restrict__ dst,
                             const float* __restrict__ a_s, const float* __restrict__ a_d,
                             float* __restrict__ ebuf, float* __restrict__ mmax,
                             int E, int EE) {
  int e = blockIdx.x * blockDim.x + threadIdx.x;
  if (e >= EE) return;
  if (e + 256 < E) { __builtin_prefetch(src + e + 256, 0, 1); __builtin_prefetch(dst + e + 256, 0, 1); }
  int s = (e < E) ? src[e] : (e - E);
  int d = (e < E) ? dst[e] : (e - E);
  float v = a_s[s] + a_d[d];
  v = (v > 0.f) ? v : NEG_SLOPE * v;
  ebuf[e] = v;
  atomicMaxF(mmax + d, v);
}

__global__ void k_edge_exp(const int* __restrict__ dst,
                           float* __restrict__ ebuf, const float* __restrict__ mmax,
                           float* __restrict__ den, int E, int EE) {
  int e = blockIdx.x * blockDim.x + threadIdx.x;
  if (e >= EE) return;
  int d = (e < E) ? dst[e] : (e - E);
  float ex = __expf(ebuf[e] - mmax[d]);
  ebuf[e] = ex;
  atomicAdd(den + d, ex);
}

// wave-per-edge: edge index loaded once (same-address broadcast), alpha computed
// once, each lane does 2 coalesced loads + 2 global_atomic_add_f32.
__global__ void k_gat_scatter(const int* __restrict__ src, const int* __restrict__ dst,
                              const float* __restrict__ ebuf, const float* __restrict__ den,
                              const float* __restrict__ h, float* __restrict__ agg,
                              int E, int EE) {
  int e    = (int)((blockIdx.x * (long)blockDim.x + threadIdx.x) >> 5);
  int lane = threadIdx.x & 31;
  if (e >= EE) return;
  int s  = (e < E) ? src[e] : (e - E);
  int dn = (e < E) ? dst[e] : (e - E);
  float alpha = ebuf[e] / den[dn];
  const float* hs = h + (size_t)s * D_HID;
  float*       ag = agg + (size_t)dn * D_HID;
  atomicAdd(ag + lane,      hs[lane]      * alpha);
  atomicAdd(ag + lane + 32, hs[lane + 32] * alpha);
}

__global__ void k_ggnn_scatter(const int* __restrict__ src, const int* __restrict__ dst,
                               const float* __restrict__ m, float* __restrict__ agg, int E) {
  int e    = (int)((blockIdx.x * (long)blockDim.x + threadIdx.x) >> 5);
  int lane = threadIdx.x & 31;
  if (e >= E) return;
  const float* ms = m   + (size_t)src[e] * D_HID;
  float*       ag = agg + (size_t)dst[e] * D_HID;
  atomicAdd(ag + lane,      ms[lane]);
  atomicAdd(ag + lane + 32, ms[lane + 32]);
}

// ------------------------------------------------------------- elementwise
__global__ void k_bias_relu(const float* __restrict__ x, const float* __restrict__ bias,
                            float* __restrict__ y, long n) {
  long i = (long)blockIdx.x * blockDim.x + threadIdx.x;
  if (i < n) y[i] = fmaxf(x[i] + bias[i & 63], 0.0f);
}

__global__ void k_gru(const float* __restrict__ gi, const float* __restrict__ gh,
                      const float* __restrict__ h_in, float* __restrict__ h_out,
                      long n64, int relu_out) {
  long i = (long)blockIdx.x * blockDim.x + threadIdx.x;
  if (i >= n64) return;
  int node = (int)(i >> 6), d = (int)(i & 63);
  size_t b = (size_t)node * 192 + d;
  float ir = gi[b], iz = gi[b + 64], in_ = gi[b + 128];
  float hr = gh[b], hz = gh[b + 64], hn  = gh[b + 128];
  float r  = 1.0f / (1.0f + __expf(-(ir + hr)));
  float z  = 1.0f / (1.0f + __expf(-(iz + hz)));
  float ng = tanhf(in_ + r * hn);
  float hv = (1.0f - z) * ng + z * h_in[i];
  if (relu_out) hv = fmaxf(hv, 0.0f);
  h_out[i] = hv;
}

// ------------------------------------------------------------------ driver
extern "C" void kernel_launch(void* const* d_in, const int* in_sizes, int n_in,
                              void* d_out, int out_size, void* d_ws, size_t ws_size,
                              hipStream_t stream) {
  const float* hidden = (const float*)d_in[1];
  const int*   ei     = (const int*)d_in[2];
  const float* gat_w  = (const float*)d_in[3];
  const float* att_s  = (const float*)d_in[4];
  const float* att_d  = (const float*)d_in[5];
  const float* gat_b  = (const float*)d_in[6];
  const float* ggnn_w = (const float*)d_in[7];
  const float* w_ih   = (const float*)d_in[8];
  const float* w_hh   = (const float*)d_in[9];
  const float* b_ih   = (const float*)d_in[10];
  const float* b_hh   = (const float*)d_in[11];

  const int N  = in_sizes[1] / D_HID;   // 100000 (multiple of 16)
  const int E  = in_sizes[2] / 2;       // 1200000
  const int EE = E + N;                 // + self loops
  const int* src = ei;
  const int* dst = ei + E;

  float* ws   = (float*)d_ws;
  float* bh   = ws; ws += (size_t)N * D_HID;   // hidden state
  float* bt   = ws; ws += (size_t)N * D_HID;   // h1 / m
  float* bagg = ws; ws += (size_t)N * D_HID;   // scatter accumulator
  float* bgi  = ws; ws += (size_t)N * 192;     // GRU input gates
  float* bgh  = ws; ws += (size_t)N * 192;     // GRU hidden gates
  float* asv  = ws; ws += N;
  float* adv  = ws; ws += N;
  float* mmax = ws; ws += N;
  float* den  = ws; ws += N;
  float* ebuf = ws; ws += EE;
  float* out  = (float*)d_out;

  auto cdiv = [](long a, long b) { return (int)((a + b - 1) / b); };
  const long N64 = (long)N * D_HID;

  // ---------------- GAT ----------------
  k_gemm_wmma<<<cdiv((long)(N / 16) * 4, 8), 256, 0, stream>>>(hidden, gat_w, nullptr, bt, N, 64, 0, 0);
  k_attn_dot<<<cdiv((long)N * 32, 256), 256, 0, stream>>>(bt, att_s, att_d, asv, adv, N);
  k_fill<<<cdiv(N, 256), 256, 0, stream>>>(mmax, -INFINITY, N);
  k_fill<<<cdiv(N, 256), 256, 0, stream>>>(den, 0.0f, N);
  k_fill<<<cdiv(N64, 256), 256, 0, stream>>>(bagg, 0.0f, N64);
  k_edge_logit<<<cdiv(EE, 256), 256, 0, stream>>>(src, dst, asv, adv, ebuf, mmax, E, EE);
  k_edge_exp<<<cdiv(EE, 256), 256, 0, stream>>>(dst, ebuf, mmax, den, E, EE);
  k_gat_scatter<<<cdiv((long)EE * 32, 256), 256, 0, stream>>>(src, dst, ebuf, den, bt, bagg, E, EE);
  k_bias_relu<<<cdiv(N64, 256), 256, 0, stream>>>(bagg, gat_b, bh, N64);

  // ---------------- GatedGraphConv (2 steps) ----------------
  for (int l = 0; l < TSTEPS; ++l) {
    k_gemm_wmma<<<cdiv((long)(N / 16) * 4, 8), 256, 0, stream>>>(bh, ggnn_w + (size_t)l * 64 * 64, nullptr, bt, N, 64, 0, 0);
    k_fill<<<cdiv(N64, 256), 256, 0, stream>>>(bagg, 0.0f, N64);
    k_ggnn_scatter<<<cdiv((long)E * 32, 256), 256, 0, stream>>>(src, dst, bt, bagg, E);
    k_gemm_wmma<<<cdiv((long)(N / 16) * 12, 8), 256, 0, stream>>>(bagg, w_ih, b_ih, bgi, N, 192, 1, 0);
    k_gemm_wmma<<<cdiv((long)(N / 16) * 12, 8), 256, 0, stream>>>(bh,   w_hh, b_hh, bgh, N, 192, 1, 0);
    float* hout = (l == TSTEPS - 1) ? out : bh;
    k_gru<<<cdiv(N64, 256), 256, 0, stream>>>(bgi, bgh, bh, hout, N64, (l == TSTEPS - 1) ? 1 : 0);
  }
}
